// SE3Transformer_28767690948907
// MI455X (gfx1250) — matrix operand order repaired and checked by reading, not compiled, not run on hardware
//
#include <hip/hip_runtime.h>
#include <cmath>
#include <cstring>

// ---------------------------------------------------------------------------
// SE(3)-Transformer forward for MI455X (gfx1250, wave32, WMMA).
// Radial MLPs run on v_wmma_f32_16x16x32_f16; graph ops are HBM-bound VALU.
// Input leaves assumed in setup_inputs() dict insertion order, recursively:
//   f, w, r, basis["di,do"] (di outer, do inner, 16),
//   params.layers[0]{attn{v(24),k(6),q(1),proj(4)},norm(8)},
//   params.layers[1]{attn{v(96),k(96),q(4),proj(4)},norm(8)},
//   params.final{r(24),si(1)}, params.fc{l1.w,l1.b,l2.w,l2.b}, src, dst
// ---------------------------------------------------------------------------

typedef __attribute__((ext_vector_type(16))) _Float16 v16h;
typedef __attribute__((ext_vector_type(8)))  _Float16 v8h;
typedef __attribute__((ext_vector_type(8)))  float    v8f;

#define TPB 256

// ---------------- WMMA fragment helpers (CDNA5 16x16x32 f16 layouts) --------

__device__ __forceinline__ v16h frag_a16(const _Float16* __restrict__ base,
                                         int stride, int lane) {
  // A 16x32 f16: lanes 0-15 row M=lane hold K{0..7,16..23}; lanes 16-31 +8.
  int row  = lane & 15;
  int koff = (lane >> 4) << 3;                   // 0 or 8
  const _Float16* p = base + row * stride + koff;
  v8h lo = *(const v8h*)(p);
  v8h hi = *(const v8h*)(p + 16);
  v16h a;
#pragma unroll
  for (int t = 0; t < 8; ++t) { a[t] = lo[t]; a[t + 8] = hi[t]; }
  return a;
}

__device__ __forceinline__ v16h frag_b16(const _Float16* __restrict__ wmat,
                                         int n0, int lane) {
  // B 32x16 f16, source stored [n][k] (k stride 32): lanes 0-15 hold
  // K{0..7,16..23} of column n, lanes 16-31 +8.
  int n  = n0 + (lane & 15);
  int kb = (lane >> 4) << 3;                     // 0 or 8
  const _Float16* p = wmat + n * 32 + kb;
  v8h lo = *(const v8h*)(p);
  v8h hi = *(const v8h*)(p + 16);
  v16h b;
#pragma unroll
  for (int t = 0; t < 8; ++t) { b[t] = lo[t]; b[t + 8] = hi[t]; }
  return b;
}

__device__ __forceinline__ v8f wmma16(const v16h a, const v16h b, const v8f c) {
  return __builtin_amdgcn_wmma_f32_16x16x32_f16(false, a, false, b,
                                                (short)0, c, false, false);
}

__device__ __forceinline__ void dfrag_bias_relu_to_lds(
    _Float16* __restrict__ sH, const v8f c, const float* __restrict__ bias,
    int n0, int lane) {
  // D f32 16x16: lanes 0-15 N=lane, M=vgpr r; lanes 16-31 N=lane-16, M=8+r.
  int col   = n0 + (lane & 15);
  int rbase = (lane >> 4) << 3;
  float bv = bias[col];
#pragma unroll
  for (int rr = 0; rr < 8; ++rr) {
    float v = fmaxf(c[rr] + bv, 0.0f);
    sH[(rbase + rr) * 32 + col] = (_Float16)v;
  }
}

// ---------------- Kernels ---------------------------------------------------

__global__ void k_build_efh(const float* __restrict__ w,
                            const float* __restrict__ r,
                            _Float16* __restrict__ efh, int E) {
  int i = blockIdx.x * blockDim.x + threadIdx.x;
  if (i >= E * 32) return;
  int e = i >> 5, j = i & 31;
  float v = 0.0f;
  if (j < 4)      v = w[e * 4 + j];
  else if (j == 4) v = r[e];
  efh[i] = (_Float16)v;
}

// Batched radial MLP: [16,5]->[16,32]->[16,32]->[16,Dout], one wave per tile.
__global__ __launch_bounds__(256) void k_radial_mlp(
    const _Float16* __restrict__ efh,
    const float* __restrict__ w1, const float* __restrict__ b1,
    const float* __restrict__ w2, const float* __restrict__ b2,
    const float* __restrict__ w3, const float* __restrict__ b3,
    float* __restrict__ Rout, int Dout, int E) {
  __shared__ __align__(16) _Float16 sB1[32 * 32];
  __shared__ __align__(16) _Float16 sB2[32 * 32];
  __shared__ __align__(16) _Float16 sB3[256 * 32];
  __shared__ __align__(16) _Float16 sH[8][16 * 32];
  __shared__ float sb1[32], sb2[32], sb3[256];

  int t = threadIdx.x;
  int Npad = (Dout + 15) & ~15;
  for (int idx = t; idx < 32 * 32; idx += TPB) {
    int n = idx >> 5, k = idx & 31;
    sB1[idx] = (_Float16)((k < 5) ? w1[n * 5 + k] : 0.0f);
    sB2[idx] = (_Float16)w2[idx];
  }
  for (int idx = t; idx < Npad * 32; idx += TPB) {
    int n = idx >> 5, k = idx & 31;
    sB3[idx] = (_Float16)((n < Dout) ? w3[n * 32 + k] : 0.0f);
  }
  if (t < 32) { sb1[t] = b1[t]; sb2[t] = b2[t]; }
  for (int idx = t; idx < Dout; idx += TPB) sb3[idx] = b3[idx];
  __syncthreads();

  int lane = t & 31, wave = t >> 5;
  int nTiles = (E + 15) >> 4;
  int tile = blockIdx.x * 8 + wave;
  if (tile >= nTiles) return;
  int e0 = tile * 16;
  _Float16* sHw = &sH[wave][0];

  // GEMM1: EF(16x32, K padded) x W1T(32x32) -> relu
  v16h a1 = frag_a16(efh + (size_t)e0 * 32, 32, lane);
#pragma unroll
  for (int nt = 0; nt < 2; ++nt) {
    v8f c = {};
    c = wmma16(a1, frag_b16(sB1, nt * 16, lane), c);
    dfrag_bias_relu_to_lds(sHw, c, sb1, nt * 16, lane);
  }
  // GEMM2: H1(16x32) x W2T(32x32) -> relu (a loaded before sH overwrite)
  v16h a2 = frag_a16(sHw, 32, lane);
#pragma unroll
  for (int nt = 0; nt < 2; ++nt) {
    v8f c = {};
    c = wmma16(a2, frag_b16(sB2, nt * 16, lane), c);
    dfrag_bias_relu_to_lds(sHw, c, sb2, nt * 16, lane);
  }
  // GEMM3: H2(16x32) x W3T(32xDout) -> global R
  v16h a3 = frag_a16(sHw, 32, lane);
  for (int nt = 0; nt < (Npad >> 4); ++nt) {
    v8f c = {};
    c = wmma16(a3, frag_b16(sB3, nt * 16, lane), c);
    int col = nt * 16 + (lane & 15);
    int rbase = (lane >> 4) << 3;
    if (col < Dout) {
      float bv = sb3[col];
#pragma unroll
      for (int rr = 0; rr < 8; ++rr) {
        int row = e0 + rbase + rr;
        if (row < E) Rout[(size_t)row * Dout + col] = c[rr] + bv;
      }
    }
  }
}

// pair message: out[e,o,p] = sum_{q,f} basis[e,p,q,f] * G[o,f,q],
//               G[o,f,q]   = sum_i R[e,o,i,f] * h_src[e,i,q]
__global__ void k_pair_msg(const float* __restrict__ Rbuf,
                           const float* __restrict__ basis,
                           const float* __restrict__ hsrc,
                           const int* __restrict__ src,
                           float* __restrict__ out,
                           int nc_in, int nc_out, int P, int Q, int F,
                           int E, int acc) {
  int e = blockIdx.x * blockDim.x + threadIdx.x;
  if (e >= E) return;
  int s = src[e];
  float h[8][7];
  for (int i = 0; i < nc_in; ++i)
    for (int q = 0; q < Q; ++q)
      h[i][q] = hsrc[((size_t)s * nc_in + i) * Q + q];
  int Dout = nc_out * nc_in * F;
  const float* Re = Rbuf + (size_t)e * Dout;
  const float* Be = basis + (size_t)e * P * Q * F;
  __builtin_prefetch(Be + (size_t)TPB * P * Q * F, 0, 1);   // next stripe
  if (P == 1 && F == 1) {
    float tq[8];
    for (int i = 0; i < nc_in; ++i) {
      float a = 0.0f;
      for (int q = 0; q < Q; ++q) a += Be[q] * h[i][q];
      tq[i] = a;
    }
    for (int o = 0; o < nc_out; ++o) {
      float a = 0.0f;
      for (int i = 0; i < nc_in; ++i) a += Re[o * nc_in + i] * tq[i];
      float* op = out + (size_t)e * nc_out + o;
      *op = acc ? (*op + a) : a;
    }
  } else {
    float G[2][7][7];                     // nc_out <= 2 on this path
    for (int o = 0; o < nc_out; ++o)
      for (int f = 0; f < F; ++f)
        for (int q = 0; q < Q; ++q) {
          float a = 0.0f;
          for (int i = 0; i < nc_in; ++i)
            a += Re[(o * nc_in + i) * F + f] * h[i][q];
          G[o][f][q] = a;
        }
    for (int o = 0; o < nc_out; ++o)
      for (int p = 0; p < P; ++p) {
        float a = 0.0f;
        for (int q = 0; q < Q; ++q)
          for (int f = 0; f < F; ++f)
            a += Be[(p * Q + q) * F + f] * G[o][f][q];
        float* op = out + ((size_t)e * nc_out + o) * P + p;
        *op = acc ? (*op + a) : a;
      }
  }
}

__global__ void k_qproj(const float* __restrict__ qw,
                        const float* __restrict__ h,
                        float* __restrict__ out, int M, int Nn) {
  int n = blockIdx.x * blockDim.x + threadIdx.x;
  if (n >= Nn) return;
  for (int o = 0; o < 2; ++o)
    for (int m = 0; m < M; ++m) {
      float a = 0.0f;
      for (int c = 0; c < 8; ++c)
        a += qw[o * 8 + c] * h[((size_t)n * 8 + c) * M + m];
      out[((size_t)n * 2 + o) * M + m] = a;
    }
}

__global__ void k_logit(const float* __restrict__ ke,
                        const float* __restrict__ qd,
                        const int* __restrict__ dst,
                        float* __restrict__ logit, int M, int E, int acc) {
  int e = blockIdx.x * blockDim.x + threadIdx.x;
  if (e >= E) return;
  int d = dst[e];
  for (int hh = 0; hh < 2; ++hh) {
    float a = 0.0f;
    for (int m = 0; m < M; ++m)
      a += ke[((size_t)e * 2 + hh) * M + m] * qd[((size_t)d * 2 + hh) * M + m];
    float* lp = logit + (size_t)e * 2 + hh;
    *lp = acc ? (*lp + a) : a;
  }
}

__device__ __forceinline__ unsigned encf(float f) {
  unsigned u = __float_as_uint(f);
  return (u & 0x80000000u) ? ~u : (u | 0x80000000u);
}
__device__ __forceinline__ float decf(unsigned u) {
  unsigned v = (u & 0x80000000u) ? (u & 0x7fffffffu) : ~u;
  return __uint_as_float(v);
}

__global__ void k_fill_f(float* p, float v, int n) {
  int i = blockIdx.x * blockDim.x + threadIdx.x;
  if (i < n) p[i] = v;
}
__global__ void k_fill_u(unsigned* p, unsigned v, int n) {
  int i = blockIdx.x * blockDim.x + threadIdx.x;
  if (i < n) p[i] = v;
}

__global__ void k_scale_segmax(float* __restrict__ logit,
                               const int* __restrict__ dst,
                               unsigned* __restrict__ m, float scale, int E) {
  int e = blockIdx.x * blockDim.x + threadIdx.x;
  if (e >= E) return;
  int d = dst[e];
  for (int hh = 0; hh < 2; ++hh) {
    float v = logit[(size_t)e * 2 + hh] * scale;
    logit[(size_t)e * 2 + hh] = v;
    atomicMax(&m[(size_t)d * 2 + hh], encf(v));
  }
}

__global__ void k_expsum(const float* __restrict__ logit,
                         const int* __restrict__ dst,
                         const unsigned* __restrict__ m,
                         float* __restrict__ a, float* __restrict__ s, int E) {
  int e = blockIdx.x * blockDim.x + threadIdx.x;
  if (e >= E) return;
  int d = dst[e];
  for (int hh = 0; hh < 2; ++hh) {
    float ex = __expf(logit[(size_t)e * 2 + hh] - decf(m[(size_t)d * 2 + hh]));
    a[(size_t)e * 2 + hh] = ex;
    atomicAdd(&s[(size_t)d * 2 + hh], ex);
  }
}

__global__ void k_attn(float* __restrict__ a, const int* __restrict__ dst,
                       const float* __restrict__ s, int E) {
  int e = blockIdx.x * blockDim.x + threadIdx.x;
  if (e >= E) return;
  int d = dst[e];
  for (int hh = 0; hh < 2; ++hh)
    a[(size_t)e * 2 + hh] /= (s[(size_t)d * 2 + hh] + 1e-9f);
}

__global__ void k_zdacc(const float* __restrict__ v, const float* __restrict__ a,
                        const int* __restrict__ dst, float* __restrict__ zd,
                        int P, int E) {
  int e = blockIdx.x * blockDim.x + threadIdx.x;
  if (e >= E) return;
  int d = dst[e];
  for (int c = 0; c < 2; ++c) {
    float av = a[(size_t)e * 2 + c];            // head == channel (NCV/NH==1)
    for (int p = 0; p < P; ++p)
      atomicAdd(&zd[((size_t)d * 2 + c) * P + p],
                v[((size_t)e * 2 + c) * P + p] * av);
  }
}

__global__ void k_proj(const float* __restrict__ proj, const float* __restrict__ zd,
                       const float* __restrict__ hprev, float* __restrict__ hout,
                       int P, int Nn) {
  int n = blockIdx.x * blockDim.x + threadIdx.x;
  if (n >= Nn) return;
  int zc = hprev ? 10 : 2;
  for (int o = 0; o < 8; ++o)
    for (int p = 0; p < P; ++p) {
      float a = 0.0f;
      for (int c = 0; c < 2; ++c)
        a += proj[o * zc + c] * zd[((size_t)n * 2 + c) * P + p];
      if (hprev) {
        for (int c = 0; c < 8; ++c)
          a += proj[o * zc + 2 + c] * hprev[((size_t)n * 8 + c) * P + p];
        a += hprev[((size_t)n * 8 + o) * P + p];   // residual
      }
      hout[((size_t)n * 8 + o) * P + p] = a;
    }
}

__global__ void k_gnorm(const float* __restrict__ wN, const float* __restrict__ bN,
                        const float* __restrict__ hin, float* __restrict__ hout,
                        int P, int Nn) {
  int n = blockIdx.x * blockDim.x + threadIdx.x;
  if (n >= Nn) return;
  float nn[8], tt[8];
  for (int c = 0; c < 8; ++c) {
    float s2 = 0.0f;
    for (int p = 0; p < P; ++p) {
      float x = hin[((size_t)n * 8 + c) * P + p];
      s2 += x * x;
    }
    nn[c] = sqrtf(s2 + 1e-12f);
  }
  for (int o = 0; o < 8; ++o) {
    float a = bN[o];
    for (int c = 0; c < 8; ++c) a += wN[o * 8 + c] * nn[c];
    tt[o] = fmaxf(a, 0.0f);
  }
  for (int o = 0; o < 8; ++o)
    for (int p = 0; p < P; ++p) {
      size_t idx = ((size_t)n * 8 + o) * P + p;
      hout[idx] = tt[o] * hin[idx] / nn[o];
    }
}

__global__ void k_cnt(const int* __restrict__ dst, float* __restrict__ cnt, int E) {
  int e = blockIdx.x * blockDim.x + threadIdx.x;
  if (e >= E) return;
  atomicAdd(&cnt[dst[e]], 1.0f);
}

__global__ void k_aggmsg(const float* __restrict__ msg, const int* __restrict__ dst,
                         float* __restrict__ agg, int E) {
  int e = blockIdx.x * blockDim.x + threadIdx.x;
  if (e >= E) return;
  int d = dst[e];
  for (int o = 0; o < 32; ++o)
    atomicAdd(&agg[(size_t)d * 32 + o], msg[(size_t)e * 32 + o]);
}

__global__ void k_final_node(const float* __restrict__ agg, const float* __restrict__ cnt,
                             const float* __restrict__ si, const float* __restrict__ h0,
                             float* __restrict__ pooled, int Nn) {
  int n = blockIdx.x * blockDim.x + threadIdx.x;
  if (n >= Nn) return;
  float c = fmaxf(cnt[n], 1.0f);
  float inv = 1.0f / (float)Nn;
  for (int o = 0; o < 32; ++o) {
    float a = agg[(size_t)n * 32 + o] / c;
    for (int ci = 0; ci < 8; ++ci) a += si[o * 8 + ci] * h0[(size_t)n * 8 + ci];
    atomicAdd(&pooled[o], a * inv);
  }
}

__global__ void k_fc(const float* __restrict__ pooled,
                     const float* __restrict__ w1, const float* __restrict__ b1,
                     const float* __restrict__ w2, const float* __restrict__ b2,
                     float* __restrict__ out) {
  __shared__ float hf[32];
  int t = threadIdx.x;
  if (t < 32) {
    float a = b1[t];
    for (int c = 0; c < 32; ++c) a += w1[t * 32 + c] * pooled[c];
    hf[t] = fmaxf(a, 0.0f);
  }
  __syncthreads();
  if (t == 0) {
    float a = b2[0];
    for (int o = 0; o < 32; ++o) a += w2[o] * hf[o];
    out[0] = a;
  }
}

// ---------------- Host ------------------------------------------------------

extern "C" void kernel_launch(void* const* d_in, const int* in_sizes, int n_in,
                              void* d_out, int out_size, void* d_ws, size_t ws_size,
                              hipStream_t stream) {
  (void)n_in; (void)out_size; (void)ws_size;
  const int Nn = in_sizes[0] / 8;        // f: [N,8,1]
  const int E  = in_sizes[1] / 4;        // w: [E,4]

  int ip = 0;
  auto F = [&](void) { return (const float*)d_in[ip++]; };

  const float* f_in = F();
  const float* w_in = F();
  const float* r_in = F();
  const float* basisP[4][4];
  for (int di = 0; di < 4; ++di)
    for (int od = 0; od < 4; ++od) basisP[di][od] = F();

  struct Rad { const float *w1, *b1, *w2, *b2, *w3, *b3; };
  struct Lin { const float *w, *b; };
  auto takeRad = [&](void) {
    Rad rr; rr.w1 = F(); rr.b1 = F(); rr.w2 = F(); rr.b2 = F();
    rr.w3 = F(); rr.b3 = F(); return rr;
  };
  auto takeLin = [&](void) { Lin l; l.w = F(); l.b = F(); return l; };

  Rad v0[4][4], k0[4][4], v1[4][4], k1[4][4], frad[4];
  const float* q0[4]; const float* q1[4];
  const float* proj0[4]; const float* proj1[4];
  Lin norm0[4], norm1[4];

  // layer 0 (din = {0})
  for (int od = 0; od < 4; ++od) v0[0][od] = takeRad();
  k0[0][0] = takeRad();
  q0[0] = F();
  for (int od = 0; od < 4; ++od) proj0[od] = F();
  for (int d = 0; d < 4; ++d) norm0[d] = takeLin();
  // layer 1 (din = {0,1,2,3})
  for (int di = 0; di < 4; ++di)
    for (int od = 0; od < 4; ++od) v1[di][od] = takeRad();
  for (int di = 0; di < 4; ++di)
    for (int od = 0; od < 4; ++od) k1[di][od] = takeRad();
  for (int d = 0; d < 4; ++d) q1[d] = F();
  for (int od = 0; od < 4; ++od) proj1[od] = F();
  for (int d = 0; d < 4; ++d) norm1[d] = takeLin();
  // final + fc
  for (int di = 0; di < 4; ++di) frad[di] = takeRad();
  const float* si = F();
  Lin fc1 = takeLin(), fc2 = takeLin();
  const int* src = (const int*)d_in[ip++];
  const int* dst = (const int*)d_in[ip++];

  // workspace carve-out
  char* ws = (char*)d_ws;
  size_t off = 0;
  auto carve = [&](size_t bytes) -> void* {
    void* p = ws + off;
    off = (off + bytes + 255) & ~(size_t)255;
    return p;
  };
  _Float16* efh = (_Float16*)carve((size_t)E * 32 * 2);
  float* Rbuf   = (float*)carve((size_t)E * 256 * 4);
  float* vbuf[4]; float* kbuf[4]; float* qb[4];
  for (int od = 0; od < 4; ++od) vbuf[od] = (float*)carve((size_t)E * 2 * (2 * od + 1) * 4);
  for (int od = 0; od < 4; ++od) kbuf[od] = (float*)carve((size_t)E * 2 * (2 * od + 1) * 4);
  for (int d = 0; d < 4; ++d)   qb[d]   = (float*)carve((size_t)Nn * 2 * (2 * d + 1) * 4);
  float*    logit = (float*)carve((size_t)E * 2 * 4);
  unsigned* mbuf  = (unsigned*)carve((size_t)Nn * 2 * 4);
  float*    sbuf  = (float*)carve((size_t)Nn * 2 * 4);
  float*    abuf  = (float*)carve((size_t)E * 2 * 4);
  float*    zd    = (float*)carve((size_t)Nn * 2 * 7 * 4);
  float* hA[4]; float* hB[4];
  for (int d = 0; d < 4; ++d) hA[d] = (float*)carve((size_t)Nn * 8 * (2 * d + 1) * 4);
  for (int d = 0; d < 4; ++d) hB[d] = (float*)carve((size_t)Nn * 8 * (2 * d + 1) * 4);
  float* msg    = (float*)carve((size_t)E * 32 * 4);
  float* cnt    = (float*)carve((size_t)Nn * 4);
  float* agg    = (float*)carve((size_t)Nn * 32 * 4);
  float* pooled = (float*)carve(32 * 4);

  const int gE  = (E + TPB - 1) / TPB;
  const int gN  = (Nn + TPB - 1) / TPB;
  const int nTiles  = (E + 15) / 16;
  const int gRad    = (nTiles + 7) / 8;

  auto fillF = [&](float* p, float v, int n) {
    k_fill_f<<<(n + TPB - 1) / TPB, TPB, 0, stream>>>(p, v, n);
  };
  auto radial = [&](const Rad& rd, int Dout) {
    k_radial_mlp<<<gRad, TPB, 0, stream>>>(efh, rd.w1, rd.b1, rd.w2, rd.b2,
                                           rd.w3, rd.b3, Rbuf, Dout, E);
  };
  auto encH = [](float f) -> unsigned {
    unsigned u; memcpy(&u, &f, 4);
    return (u & 0x80000000u) ? ~u : (u | 0x80000000u);
  };

  k_build_efh<<<(E * 32 + TPB - 1) / TPB, TPB, 0, stream>>>(w_in, r_in, efh, E);

  auto run_layer = [&](int ndin, Rad (*vr)[4], Rad (*kr)[4],
                       const float* const* qw, const float* const* projw,
                       const Lin* nrm, const float* const* hin, float* const* hout) {
    // value messages: v[od] = sum_di pair_msg
    for (int od = 0; od < 4; ++od)
      for (int di = 0; di < ndin; ++di) {
        int nf = 2 * (di < od ? di : od) + 1;
        radial(vr[di][od], 2 * 8 * nf);
        k_pair_msg<<<gE, TPB, 0, stream>>>(Rbuf, basisP[di][od], hin[di], src,
                                           vbuf[od], 8, 2, 2 * od + 1,
                                           2 * di + 1, nf, E, di > 0);
      }
    // key messages: ke[od] for od in din
    for (int od = 0; od < ndin; ++od)
      for (int di = 0; di < ndin; ++di) {
        int nf = 2 * (di < od ? di : od) + 1;
        radial(kr[di][od], 2 * 8 * nf);
        k_pair_msg<<<gE, TPB, 0, stream>>>(Rbuf, basisP[di][od], hin[di], src,
                                           kbuf[od], 8, 2, 2 * od + 1,
                                           2 * di + 1, nf, E, di > 0);
      }
    // queries + logits
    for (int d = 0; d < ndin; ++d)
      k_qproj<<<gN, TPB, 0, stream>>>(qw[d], hin[d], qb[d], 2 * d + 1, Nn);
    for (int d = 0; d < ndin; ++d)
      k_logit<<<gE, TPB, 0, stream>>>(kbuf[d], qb[d], dst, logit, 2 * d + 1, E, d > 0);
    int dk = 0;
    for (int d = 0; d < ndin; ++d) dk += 2 * d + 1;
    float scale = 1.0f / sqrtf((float)dk);
    k_fill_u<<<(Nn * 2 + TPB - 1) / TPB, TPB, 0, stream>>>(mbuf, encH(-3.0e38f), Nn * 2);
    k_scale_segmax<<<gE, TPB, 0, stream>>>(logit, dst, mbuf, scale, E);
    fillF(sbuf, 0.0f, Nn * 2);
    k_expsum<<<gE, TPB, 0, stream>>>(logit, dst, mbuf, abuf, sbuf, E);
    k_attn<<<gE, TPB, 0, stream>>>(abuf, dst, sbuf, E);
    // aggregate, project, residual, norm
    for (int od = 0; od < 4; ++od) {
      int P = 2 * od + 1;
      fillF(zd, 0.0f, Nn * 2 * P);
      k_zdacc<<<gE, TPB, 0, stream>>>(vbuf[od], abuf, dst, zd, P, E);
      const float* hp = (od < ndin) ? hin[od] : nullptr;
      k_proj<<<gN, TPB, 0, stream>>>(projw[od], zd, hp, hout[od], P, Nn);
    }
    for (int d = 0; d < 4; ++d)
      k_gnorm<<<gN, TPB, 0, stream>>>(nrm[d].w, nrm[d].b, hout[d], hout[d],
                                      2 * d + 1, Nn);
  };

  const float* hin0[4] = { f_in, nullptr, nullptr, nullptr };
  run_layer(1, v0, k0, q0, proj0, norm0, hin0, hA);
  const float* hin1[4] = { hA[0], hA[1], hA[2], hA[3] };
  run_layer(4, v1, k1, q1, proj1, norm1, hin1, hB);

  // final degree-0 conv: msg[e,32] = sum_di pair_msg(frad[di], basis[di,0], hB[di])
  for (int di = 0; di < 4; ++di) {
    radial(frad[di], 32 * 8 * 1);
    k_pair_msg<<<gE, TPB, 0, stream>>>(Rbuf, basisP[di][0], hB[di], src, msg,
                                       8, 32, 1, 2 * di + 1, 1, E, di > 0);
  }
  fillF(cnt, 0.0f, Nn);
  k_cnt<<<gE, TPB, 0, stream>>>(dst, cnt, E);
  fillF(agg, 0.0f, Nn * 32);
  k_aggmsg<<<gE, TPB, 0, stream>>>(msg, dst, agg, E);
  fillF(pooled, 0.0f, 32);
  k_final_node<<<gN, TPB, 0, stream>>>(agg, cnt, si, hB[0], pooled, Nn);
  k_fc<<<1, 32, 0, stream>>>(pooled, fc1.w, fc1.b, fc2.w, fc2.b, (float*)d_out);
}
